// AttentionBlock_19267223290170
// MI455X (gfx1250) — matrix operand order repaired and checked by reading, not compiled, not run on hardware
//
#include <hip/hip_runtime.h>
#include <hip/hip_bf16.h>

// Problem constants (match reference)
#define B_    2
#define L_    2048
#define DIM_  1024
#define H_    16
#define D_    64
#define MTOT  (B_*L_)      // 4096
#define NQKV  (3*H_*D_)    // 3072
#define SCALE 0.125f       // 1/sqrt(64)

typedef __attribute__((ext_vector_type(16))) __bf16 v16bf;
typedef __attribute__((ext_vector_type(8)))  float  v8f;

union Frag { v16bf v; unsigned u[8]; };

__device__ __forceinline__ unsigned short f32_to_bf16(float f) {
  unsigned u = __float_as_uint(f);
  u += 0x7FFFu + ((u >> 16) & 1u);          // round-to-nearest-even
  return (unsigned short)(u >> 16);
}
__device__ __forceinline__ float bf16_to_f32(unsigned short h) {
  return __uint_as_float(((unsigned)h) << 16);
}

__device__ __forceinline__ v8f wmma_bf16(v16bf a, v16bf b, v8f c) {
  // (neg_a, A, neg_b, B, c_mod, C, reuse_a, reuse_b)
  return __builtin_amdgcn_wmma_f32_16x16x32_bf16(false, a, false, b, (short)0, c,
                                                 false, false);
}

// ---- fragment gathers (LDS or global; all accesses dword-aligned and the
// compiler coalesces the pair-loads into ds_load_b128) ----
// A operand: 16x32 bf16. lanes 0-15: K 0-15, lanes 16-31: K 16-31.
__device__ __forceinline__ v16bf frag_A(const unsigned short* p0, int lda, int lane) {
  Frag f;
  const int m = lane & 15, half = lane >> 4;
  const unsigned short* p = p0 + m * lda + half * 16;
  #pragma unroll
  for (int v = 0; v < 8; ++v) {
    const int kb = v * 2;
    f.u[v] = *(const unsigned*)(p + kb);
  }
  return f.v;
}
// B operand: 32x16 bf16 stored as [N][K]. lanes 0-15: K 0-15, lanes 16-31: K 16-31.
__device__ __forceinline__ v16bf frag_B(const unsigned short* p0, int lda, int lane) {
  Frag f;
  const int n = lane & 15, half = lane >> 4;
  const unsigned short* p = p0 + n * lda + half * 16;
  #pragma unroll
  for (int v = 0; v < 8; ++v) f.u[v] = *(const unsigned*)(p + v * 2);
  return f.v;
}

// ===================== fp32 -> bf16 convert =====================
__global__ __launch_bounds__(256) void cvt_bf16_kernel(const float* __restrict__ src,
                                                       unsigned short* __restrict__ dst,
                                                       int n) {
  int i = blockIdx.x * 256 + threadIdx.x;
  if (i < n) dst[i] = f32_to_bf16(src[i]);
}

// ===================== shared GEMM mainloop (double-buffered) =====================
// 128x128 block tile, BK=32, 8 waves in 2(M)x4(N) grid, wave tile 64x32.
// As/Bs are ping-pong pairs: 2 x [128 x 40] bf16 each.
__device__ __forceinline__ void gemm_mainloop(const unsigned short* __restrict__ Ag,
                                              const unsigned short* __restrict__ Bg,
                                              int K, unsigned short* As,
                                              unsigned short* Bs, v8f acc[4][2]) {
  const int tid  = threadIdx.x;
  const int lane = tid & 31;
  const int wave = tid >> 5;
  const int wm = (wave >> 2) * 64;
  const int wn = (wave & 3) * 32;

  // two 16B chunks per thread per matrix
  const int c0 = tid, c1 = tid + 256;
  const int r0 = c0 >> 2, col0 = (c0 & 3) * 8;
  const int r1 = c1 >> 2, col1 = (c1 & 3) * 8;

  // stage tile 0 into buffer 0
  {
    uint4 a0 = *(const uint4*)(Ag + (size_t)r0 * K + col0);
    uint4 a1 = *(const uint4*)(Ag + (size_t)r1 * K + col1);
    uint4 b0 = *(const uint4*)(Bg + (size_t)r0 * K + col0);
    uint4 b1 = *(const uint4*)(Bg + (size_t)r1 * K + col1);
    *(uint4*)(As + r0 * 40 + col0) = a0;
    *(uint4*)(As + r1 * 40 + col1) = a1;
    *(uint4*)(Bs + r0 * 40 + col0) = b0;
    *(uint4*)(Bs + r1 * 40 + col1) = b1;
  }
  __syncthreads();

  int cur = 0;
  for (int k0 = 0; k0 < K; k0 += 32) {
    const int nxt = k0 + 32;
    uint4 na0, na1, nb0, nb1;
    if (nxt < K) {   // prefetch next tile into registers (overlaps compute)
      na0 = *(const uint4*)(Ag + (size_t)r0 * K + nxt + col0);
      na1 = *(const uint4*)(Ag + (size_t)r1 * K + nxt + col1);
      nb0 = *(const uint4*)(Bg + (size_t)r0 * K + nxt + col0);
      nb1 = *(const uint4*)(Bg + (size_t)r1 * K + nxt + col1);
    }
    const unsigned short* Ac = As + cur * (128 * 40);
    const unsigned short* Bc = Bs + cur * (128 * 40);
    v16bf b0 = frag_B(Bc + wn * 40, 40, lane);
    v16bf b1 = frag_B(Bc + (wn + 16) * 40, 40, lane);
    #pragma unroll
    for (int mt = 0; mt < 4; ++mt) {
      v16bf a = frag_A(Ac + (wm + mt * 16) * 40, 40, lane);
      acc[mt][0] = wmma_bf16(a, b0, acc[mt][0]);
      acc[mt][1] = wmma_bf16(a, b1, acc[mt][1]);
    }
    if (nxt < K) {   // store prefetched tile into the other buffer
      unsigned short* An = As + (cur ^ 1) * (128 * 40);
      unsigned short* Bn = Bs + (cur ^ 1) * (128 * 40);
      *(uint4*)(An + r0 * 40 + col0) = na0;
      *(uint4*)(An + r1 * 40 + col1) = na1;
      *(uint4*)(Bn + r0 * 40 + col0) = nb0;
      *(uint4*)(Bn + r1 * 40 + col1) = nb1;
      __syncthreads();            // single barrier per k-step
      cur ^= 1;
    }
  }
}

// ===================== QKV projection =====================
// qkv = x @ Wqkv^T + bqkv ; scatter to Q/K/V [B,H,L,D] bf16
__global__ __launch_bounds__(256) void qkv_gemm_kernel(
    const unsigned short* __restrict__ xb, const unsigned short* __restrict__ Wb,
    const float* __restrict__ bqkv, unsigned short* __restrict__ Qr,
    unsigned short* __restrict__ Kr, unsigned short* __restrict__ Vr) {
  __shared__ __align__(16) unsigned short As[2 * 128 * 40];
  __shared__ __align__(16) unsigned short Bs[2 * 128 * 40];
  const int blockM = blockIdx.x * 128;
  const int blockN = blockIdx.y * 128;
  v8f acc[4][2];
  const v8f z = {0.f, 0.f, 0.f, 0.f, 0.f, 0.f, 0.f, 0.f};
  #pragma unroll
  for (int i = 0; i < 4; ++i) { acc[i][0] = z; acc[i][1] = z; }

  gemm_mainloop(xb + (size_t)blockM * DIM_, Wb + (size_t)blockN * DIM_, DIM_, As, Bs, acc);

  const int lane = threadIdx.x & 31, wave = threadIdx.x >> 5;
  const int wm = (wave >> 2) * 64, wn = (wave & 3) * 32;
  const int half = lane >> 4, nl = lane & 15;
  #pragma unroll
  for (int mt = 0; mt < 4; ++mt) {
    const int mBase = blockM + wm + mt * 16 + 8 * half;
    #pragma unroll
    for (int nt = 0; nt < 2; ++nt) {
      const int nG = blockN + wn + nt * 16 + nl;
      const float bias = bqkv[nG];
      const int which = nG >> 10, rem = nG & 1023;
      const int h = rem >> 6, d = rem & 63;
      unsigned short* dst = (which == 0) ? Qr : ((which == 1) ? Kr : Vr);
      #pragma unroll
      for (int r = 0; r < 8; ++r) {
        const int mg = mBase + r;
        const int b = mg >> 11, l = mg & 2047;
        dst[(size_t)((((b << 4) + h) << 11) + l) * 64 + d] =
            f32_to_bf16(acc[mt][nt][r] + bias);
      }
    }
  }
}

// ===================== RoPE (in-place on Q,K) =====================
__global__ __launch_bounds__(256) void rope_kernel(unsigned short* __restrict__ Q,
                                                   unsigned short* __restrict__ K) {
  const int idx = blockIdx.x * 256 + threadIdx.x;   // B*H*L*32 threads
  if (idx >= B_ * H_ * L_ * 32) return;
  const int j  = idx & 31;
  const int l  = (idx >> 5) & 2047;
  const int bh = idx >> 16;
  // theta = l * 10000^(-j/32)
  const float theta = (float)l * __expf(-(float)j * 0.28782313662425572f);
  float sn, cs;
  __sincosf(theta, &sn, &cs);
  const size_t base = ((size_t)bh * L_ + l) * 64;
  {
    float a = bf16_to_f32(Q[base + j]), b = bf16_to_f32(Q[base + j + 32]);
    Q[base + j]      = f32_to_bf16(a * cs - b * sn);
    Q[base + j + 32] = f32_to_bf16(b * cs + a * sn);
  }
  {
    float a = bf16_to_f32(K[base + j]), b = bf16_to_f32(K[base + j + 32]);
    K[base + j]      = f32_to_bf16(a * cs - b * sn);
    K[base + j + 32] = f32_to_bf16(b * cs + a * sn);
  }
}

// ===================== flash attention (double-buffered K/V stage) ==============
// grid: (L/128, B*H). 8 waves; each wave owns a 16-row query tile.
__global__ __launch_bounds__(256) void flash_attn_kernel(
    const unsigned short* __restrict__ Q, const unsigned short* __restrict__ Kg,
    const unsigned short* __restrict__ Vg, unsigned short* __restrict__ Ob) {
  __shared__ __align__(16) unsigned short Ks[2 * 64 * 72];     // [t][d]
  __shared__ __align__(16) unsigned short Vt[2 * 64 * 68];     // [d][t] (transposed)
  __shared__ __align__(16) unsigned short Ps[8 * 16 * 66];     // per-wave P tile

  const int tid = threadIdx.x, lane = tid & 31, wave = tid >> 5;
  const int half = lane >> 4, nl = lane & 15;
  const int bh = blockIdx.y;
  const int b = bh >> 4, h = bh & 15;
  const int qRow = blockIdx.x * 128 + wave * 16;

  const unsigned short* Qp = Q  + (size_t)bh * L_ * 64;
  const unsigned short* Kp = Kg + (size_t)bh * L_ * 64;
  const unsigned short* Vp = Vg + (size_t)bh * L_ * 64;

  // staging chunk coords: two 16B chunks per thread
  const int c0 = tid, c1 = tid + 256;
  const int t_0 = c0 >> 3, dd0 = (c0 & 7) * 8;
  const int t_1 = c1 >> 3, dd1 = (c1 & 7) * 8;

  // Q fragments (2 K-chunks of 32), loaded once from global
  Frag qf[2];
  #pragma unroll
  for (int c = 0; c < 2; ++c) qf[c].v = frag_A(Qp + (size_t)qRow * 64 + c * 32, 64, lane);

  const v8f z = {0.f, 0.f, 0.f, 0.f, 0.f, 0.f, 0.f, 0.f};
  v8f o[4] = {z, z, z, z};
  v8f mrun, lrun;
  #pragma unroll
  for (int r = 0; r < 8; ++r) { mrun[r] = -1e30f; lrun[r] = 0.f; }

  unsigned short* pw = Ps + wave * 16 * 66;

  // stage block 0 into buffer 0
  {
    uint4 k0v = *(const uint4*)(Kp + (size_t)t_0 * 64 + dd0);
    uint4 k1v = *(const uint4*)(Kp + (size_t)t_1 * 64 + dd1);
    uint4 v0v = *(const uint4*)(Vp + (size_t)t_0 * 64 + dd0);
    uint4 v1v = *(const uint4*)(Vp + (size_t)t_1 * 64 + dd1);
    *(uint4*)(Ks + t_0 * 72 + dd0) = k0v;
    *(uint4*)(Ks + t_1 * 72 + dd1) = k1v;
    const unsigned short* s0 = (const unsigned short*)&v0v;
    const unsigned short* s1 = (const unsigned short*)&v1v;
    #pragma unroll
    for (int j = 0; j < 8; ++j) {
      Vt[(dd0 + j) * 68 + t_0] = s0[j];
      Vt[(dd1 + j) * 68 + t_1] = s1[j];
    }
  }
  __syncthreads();

  int cur = 0;
  for (int t0 = 0; t0 < L_; t0 += 64) {
    const int nt0 = t0 + 64;
    uint4 nk0, nk1, nv0, nv1;
    if (nt0 < L_) {   // prefetch next K/V block into registers
      nk0 = *(const uint4*)(Kp + (size_t)(nt0 + t_0) * 64 + dd0);
      nk1 = *(const uint4*)(Kp + (size_t)(nt0 + t_1) * 64 + dd1);
      nv0 = *(const uint4*)(Vp + (size_t)(nt0 + t_0) * 64 + dd0);
      nv1 = *(const uint4*)(Vp + (size_t)(nt0 + t_1) * 64 + dd1);
    }
    const unsigned short* KsC = Ks + cur * (64 * 72);
    const unsigned short* VtC = Vt + cur * (64 * 68);

    // S = Q @ K^T  (4 key-subtiles of 16)
    v8f s[4];
    #pragma unroll
    for (int nt = 0; nt < 4; ++nt) {
      s[nt] = z;
      #pragma unroll
      for (int c = 0; c < 2; ++c) {
        v16bf bf = frag_B(KsC + (nt * 16) * 72 + c * 32, 72, lane);
        s[nt] = wmma_bf16(qf[c].v, bf, s[nt]);
      }
      #pragma unroll
      for (int r = 0; r < 8; ++r) s[nt][r] *= SCALE;
    }

    // online softmax (row = r + 8*half, cols striped across 16 lanes)
    v8f rmax;
    #pragma unroll
    for (int r = 0; r < 8; ++r)
      rmax[r] = fmaxf(fmaxf(s[0][r], s[1][r]), fmaxf(s[2][r], s[3][r]));
    #pragma unroll
    for (int mask = 1; mask <= 8; mask <<= 1)
      #pragma unroll
      for (int r = 0; r < 8; ++r)
        rmax[r] = fmaxf(rmax[r], __shfl_xor(rmax[r], mask, 32));

    v8f fac, rsum;
    #pragma unroll
    for (int r = 0; r < 8; ++r) {
      float mn = fmaxf(mrun[r], rmax[r]);
      fac[r] = __expf(mrun[r] - mn);
      mrun[r] = mn;
      rsum[r] = 0.f;
    }
    #pragma unroll
    for (int nt = 0; nt < 4; ++nt)
      #pragma unroll
      for (int r = 0; r < 8; ++r) {
        float p = __expf(s[nt][r] - mrun[r]);
        s[nt][r] = p;
        rsum[r] += p;
      }
    #pragma unroll
    for (int mask = 1; mask <= 8; mask <<= 1)
      #pragma unroll
      for (int r = 0; r < 8; ++r) rsum[r] += __shfl_xor(rsum[r], mask, 32);
    #pragma unroll
    for (int r = 0; r < 8; ++r) lrun[r] = lrun[r] * fac[r] + rsum[r];
    #pragma unroll
    for (int dt = 0; dt < 4; ++dt)
      #pragma unroll
      for (int r = 0; r < 8; ++r) o[dt][r] *= fac[r];

    // re-layout P (C layout -> A layout) through per-wave LDS; same-wave LDS
    // ops are in-order so no barrier needed.
    #pragma unroll
    for (int nt = 0; nt < 4; ++nt)
      #pragma unroll
      for (int r = 0; r < 8; ++r)
        pw[(r + 8 * half) * 66 + nt * 16 + nl] = f32_to_bf16(s[nt][r]);

    // O += P @ V
    #pragma unroll
    for (int c = 0; c < 2; ++c) {
      v16bf pf = frag_A(pw + c * 32, 66, lane);
      #pragma unroll
      for (int dt = 0; dt < 4; ++dt) {
        v16bf vf = frag_B(VtC + (dt * 16) * 68 + c * 32, 68, lane);
        o[dt] = wmma_bf16(pf, vf, o[dt]);
      }
    }

    if (nt0 < L_) {   // commit prefetched block into the other buffer
      unsigned short* KsN = Ks + (cur ^ 1) * (64 * 72);
      unsigned short* VtN = Vt + (cur ^ 1) * (64 * 68);
      *(uint4*)(KsN + t_0 * 72 + dd0) = nk0;
      *(uint4*)(KsN + t_1 * 72 + dd1) = nk1;
      const unsigned short* s0 = (const unsigned short*)&nv0;
      const unsigned short* s1 = (const unsigned short*)&nv1;
      #pragma unroll
      for (int j = 0; j < 8; ++j) {
        VtN[(dd0 + j) * 68 + t_0] = s0[j];
        VtN[(dd1 + j) * 68 + t_1] = s1[j];
      }
      __syncthreads();          // single barrier per key-block
      cur ^= 1;
    }
  }

  // normalize and store to Ob [B*L, H*D] bf16
  v8f inv;
  #pragma unroll
  for (int r = 0; r < 8; ++r) inv[r] = 1.0f / lrun[r];
  const int qmBase = qRow + 8 * half;
  #pragma unroll
  for (int dt = 0; dt < 4; ++dt) {
    const int col = h * 64 + dt * 16 + nl;
    #pragma unroll
    for (int r = 0; r < 8; ++r) {
      const int q = qmBase + r;
      Ob[((size_t)(b * L_ + q)) * (H_ * D_) + col] = f32_to_bf16(o[dt][r] * inv[r]);
    }
  }
}

// ===================== output projection =====================
__global__ __launch_bounds__(256) void proj_gemm_kernel(
    const unsigned short* __restrict__ Ab, const unsigned short* __restrict__ Wb,
    const float* __restrict__ bproj, float* __restrict__ out) {
  __shared__ __align__(16) unsigned short As[2 * 128 * 40];
  __shared__ __align__(16) unsigned short Bs[2 * 128 * 40];
  const int blockM = blockIdx.x * 128;
  const int blockN = blockIdx.y * 128;
  v8f acc[4][2];
  const v8f z = {0.f, 0.f, 0.f, 0.f, 0.f, 0.f, 0.f, 0.f};
  #pragma unroll
  for (int i = 0; i < 4; ++i) { acc[i][0] = z; acc[i][1] = z; }

  gemm_mainloop(Ab + (size_t)blockM * (H_ * D_), Wb + (size_t)blockN * (H_ * D_),
                H_ * D_, As, Bs, acc);

  const int lane = threadIdx.x & 31, wave = threadIdx.x >> 5;
  const int wm = (wave >> 2) * 64, wn = (wave & 3) * 32;
  const int half = lane >> 4, nl = lane & 15;
  #pragma unroll
  for (int mt = 0; mt < 4; ++mt) {
    const int mBase = blockM + wm + mt * 16 + 8 * half;
    #pragma unroll
    for (int nt = 0; nt < 2; ++nt) {
      const int nG = blockN + wn + nt * 16 + nl;
      const float bias = bproj[nG];
      #pragma unroll
      for (int r = 0; r < 8; ++r)
        out[(size_t)(mBase + r) * DIM_ + nG] = acc[mt][nt][r] + bias;
    }
  }
}

// ===================== host launcher =====================
extern "C" void kernel_launch(void* const* d_in, const int* in_sizes, int n_in,
                              void* d_out, int out_size, void* d_ws, size_t ws_size,
                              hipStream_t stream) {
  const float* x     = (const float*)d_in[0];
  const float* Wqkv  = (const float*)d_in[1];
  const float* bqkv  = (const float*)d_in[2];
  const float* Wproj = (const float*)d_in[3];
  const float* bproj = (const float*)d_in[4];
  float* out = (float*)d_out;

  char* ws = (char*)d_ws;
  const size_t nX   = (size_t)MTOT * DIM_;        // 4,194,304
  const size_t nWq  = (size_t)NQKV * DIM_;        // 3,145,728
  const size_t nWp  = (size_t)DIM_ * (H_ * D_);   // 1,048,576
  const size_t nQKV = (size_t)B_ * H_ * L_ * D_;  // 4,194,304

  size_t off = 0;
  unsigned short* xb  = (unsigned short*)(ws + off); off += nX   * 2;
  unsigned short* Wqb = (unsigned short*)(ws + off); off += nWq  * 2;
  unsigned short* Wpb = (unsigned short*)(ws + off); off += nWp  * 2;
  unsigned short* Qr  = (unsigned short*)(ws + off); off += nQKV * 2;
  unsigned short* Kr  = (unsigned short*)(ws + off); off += nQKV * 2;
  unsigned short* Vr  = (unsigned short*)(ws + off); off += nQKV * 2;
  unsigned short* Ob  = (unsigned short*)(ws + off); off += nX   * 2;
  (void)ws_size; (void)n_in; (void)in_sizes; (void)out_size;

  cvt_bf16_kernel<<<(nX  + 255) / 256, 256, 0, stream>>>(x, xb, (int)nX);
  cvt_bf16_kernel<<<(nWq + 255) / 256, 256, 0, stream>>>(Wqkv, Wqb, (int)nWq);
  cvt_bf16_kernel<<<(nWp + 255) / 256, 256, 0, stream>>>(Wproj, Wpb, (int)nWp);

  qkv_gemm_kernel<<<dim3(MTOT / 128, NQKV / 128), 256, 0, stream>>>(
      xb, Wqb, bqkv, Qr, Kr, Vr);

  rope_kernel<<<(B_ * H_ * L_ * 32) / 256, 256, 0, stream>>>(Qr, Kr);

  flash_attn_kernel<<<dim3(L_ / 128, B_ * H_), 256, 0, stream>>>(Qr, Kr, Vr, Ob);

  proj_gemm_kernel<<<dim3(MTOT / 128, DIM_ / 128), 256, 0, stream>>>(
      Ob, Wpb, bproj, out);
}
